// ConvergeLoss_11261404250302
// MI455X (gfx1250) — matrix operand order repaired and checked by reading, not compiled, not run on hardware
//
#include <hip/hip_runtime.h>
#include <hip/hip_bf16.h>

typedef __attribute__((ext_vector_type(16))) _Float16 v16h;
typedef __attribute__((ext_vector_type(8)))  float    v8f;
typedef __attribute__((ext_vector_type(4)))  float    v4f;

// ---------------- workspace layout (floats) ----------------
#define HS_OFF   0          // 17 * 8 * 9 = 1224 homography entries
#define S2_OFF   1224       // 4
#define D2_OFF   1228       // 4
#define PART_OFF 1232       // 17 * 512 = 8704 block partial sums

#define TILE_H 8
#define TILE_W 16
#define HALO_H 10
#define HALO_W 18
#define PLANE   (HALO_H * HALO_W)   // 180 halo positions
#define CSTRIDE 72                  // channel-run stride per position (64 + 8 pad)

// =====================================================================
// Kernel A: calc_fmap via per-pixel 9x64 Gram with v_wmma_f32_16x16x32_f16
// LDS layout: xs[pos][c] (channel-major) so WMMA fragments load as b128.
// =====================================================================
__global__ __launch_bounds__(256)
void fmap_kernel(const float* __restrict__ tf, const float* __restrict__ inp,
                 float* __restrict__ out_t, float* __restrict__ out_i) {
    __shared__ float xs[PLANE * CSTRIDE];   // 51840 B
    __shared__ float mu[PLANE];             // per-position channel mean

    const int tilesPerPlane = (128 / TILE_H) * (128 / TILE_W); // 128
    int blk = blockIdx.x;
    int img = blk / (8 * tilesPerPlane);
    int rem = blk % (8 * tilesPerPlane);
    int b   = rem / tilesPerPlane;
    int t   = rem % tilesPerPlane;
    int ty  = t / (128 / TILE_W);
    int tx  = t % (128 / TILE_W);
    int y0  = ty * TILE_H, x0 = tx * TILE_W;

    const float* src = (img == 0 ? tf : inp) + (size_t)b * 64 * 128 * 128;
    float*       dst = (img == 0 ? out_t : out_i) + b * 128 * 128;

    int tid = threadIdx.x;

    // ---- stage tile: global (coalesced, x fastest) -> LDS channel-major ----
    for (int idx = tid; idx < 64 * PLANE; idx += 256) {
        int c   = idx / PLANE;
        int pos = idx % PLANE;
        int y = pos / HALO_W, x = pos % HALO_W;
        int gy = y0 - 1 + y, gx = x0 - 1 + x;
        float v = 0.0f;
        if (gy >= 0 && gy < 128 && gx >= 0 && gx < 128)
            v = src[(size_t)c * 128 * 128 + gy * 128 + gx];
        xs[pos * CSTRIDE + c] = v;
    }
    __syncthreads();

    // ---- per-position channel mean (contiguous 64 floats -> b128 loads) ----
    for (int p = tid; p < PLANE; p += 256) {
        const float* cp = &xs[p * CSTRIDE];
        float s = 0.0f;
#pragma unroll
        for (int c = 0; c < 64; c += 4) {
            v4f v = *(const v4f*)(cp + c);
            s += v[0] + v[1] + v[2] + v[3];
        }
        mu[p] = s * (1.0f / 64.0f);
    }
    __syncthreads();

    int wv   = tid >> 5;
    int lane = tid & 31;
    int halfSel = (lane >= 16) ? 1 : 0;
    int tap  = lane & 15;                 // M (for A) == N (for B) == tap index
    bool tapValid = (tap < 9);
    float vmask = tapValid ? 1.0f : 0.0f; // zero pad rows/cols via VALU, not EXEC
    int tdy = tapValid ? tap / 3 : 0;
    int tdx = tapValid ? tap % 3 : 0;
    // channel run bases within a 32-wide K chunk (contiguous K = 16h + e, same for A and B)
    int cA0 = (halfSel << 4);             // A halves 0..7  -> K = 16h + e
    int cA1 = 8 + (halfSel << 4);         // A halves 8..15 -> K = 16h + e
    int cB  = (halfSel << 4);             // B halves       -> K = 16h + e

    // 8 waves x 16 pixels = 128 pixels = full tile
    for (int i = 0; i < 16; ++i) {
        int p  = wv * 16 + i;
        int py = p / TILE_W, px = p % TILE_W;
        int pos = (py + tdy) * HALO_W + (px + tdx);
        const float* posp = &xs[pos * CSTRIDE];
        float m = mu[pos];

        v8f acc = {};
#pragma unroll
        for (int q = 0; q < 2; ++q) {     // K = 64 -> two chunks of 32
            const float* cp = posp + q * 32;
            v4f a0 = *(const v4f*)(cp + cA0);
            v4f a1 = *(const v4f*)(cp + cA0 + 4);
            v4f a2 = *(const v4f*)(cp + cA1);
            v4f a3 = *(const v4f*)(cp + cA1 + 4);
            v4f b0 = *(const v4f*)(cp + cB);
            v4f b1 = *(const v4f*)(cp + cB + 4);
            v4f b2 = *(const v4f*)(cp + cB + 8);
            v4f b3 = *(const v4f*)(cp + cB + 12);
            v16h A, B;
#pragma unroll
            for (int e = 0; e < 4; ++e) {
                A[e]      = (_Float16)((a0[e] - m) * vmask);
                A[e + 4]  = (_Float16)((a1[e] - m) * vmask);
                A[e + 8]  = (_Float16)((a2[e] - m) * vmask);
                A[e + 12] = (_Float16)((a3[e] - m) * vmask);
                B[e]      = (_Float16)((b0[e] - m) * vmask);
                B[e + 4]  = (_Float16)((b1[e] - m) * vmask);
                B[e + 8]  = (_Float16)((b2[e] - m) * vmask);
                B[e + 12] = (_Float16)((b3[e] - m) * vmask);
            }
            acc = __builtin_amdgcn_wmma_f32_16x16x32_f16(
                false, A, false, B, (short)0, acc, false, false);
        }

        // D layout: VGPR r, lane l<16 -> G[r,l]; lane l>=16 -> G[r+8,l-16].
        // G symmetric => row_sum[n] = col_sum[n].
        float partial = 0.0f;
#pragma unroll
        for (int r = 0; r < 8; ++r) partial += acc[r];
        float cs = partial + __shfl_xor(partial, 16, 32); // rows 0..7 + row 8

        // trace: lane k<8 holds G[k,k] in acc[k]; lane 24 holds G[8,8] in acc[0]
        float dg = 0.0f;
#pragma unroll
        for (int r = 0; r < 8; ++r) dg = (lane == r) ? acc[r] : dg;
        dg = (lane == 24) ? acc[0] : dg;
        float tr = dg;
#pragma unroll
        for (int off = 16; off >= 1; off >>= 1) tr += __shfl_xor(tr, off, 32);

        float csMax = tapValid ? cs : -3.4e38f;
        float csMin = tapValid ? cs :  3.4e38f;
#pragma unroll
        for (int off = 16; off >= 1; off >>= 1) {
            csMax = fmaxf(csMax, __shfl_xor(csMax, off, 32));
            csMin = fminf(csMin, __shfl_xor(csMin, off, 32));
        }

        float mimic = (csMax + csMin) * 0.5f / tr;
        if (lane == 0) dst[(y0 + py) * 128 + (x0 + px)] = mimic;
    }
}

// =====================================================================
// Kernel B: JAX threefry2x32 RNG -> lambdas, s2/d2, and 17 x 8 H matrices
// =====================================================================
__device__ inline void tf2x32(unsigned k0, unsigned k1, unsigned& x0, unsigned& x1) {
    unsigned ks0 = k0, ks1 = k1, ks2 = k0 ^ k1 ^ 0x1BD11BDAu;
    const int r0[4] = {13, 15, 26, 6};
    const int r1[4] = {17, 29, 16, 24};
    x0 += ks0; x1 += ks1;
#define TF_R(rr) { for (int i_ = 0; i_ < 4; ++i_) { x0 += x1; \
    x1 = (x1 << rr[i_]) | (x1 >> (32 - rr[i_])); x1 ^= x0; } }
    TF_R(r0); x0 += ks1; x1 += ks2 + 1u;
    TF_R(r1); x0 += ks2; x1 += ks0 + 2u;
    TF_R(r0); x0 += ks0; x1 += ks1 + 3u;
    TF_R(r1); x0 += ks1; x1 += ks2 + 4u;
    TF_R(r0); x0 += ks2; x1 += ks0 + 5u;
#undef TF_R
}

__global__ void setup_kernel(const float* __restrict__ Hgt, float* __restrict__ ws) {
    __shared__ float lam[4][9];
    __shared__ float s2s[4];
    if (threadIdx.x == 0) {
        for (int n = 0; n < 4; ++n) {
            // fold_in(key(42), n): key=(0,42), count=(0,n) -> new key
            unsigned a = 0u, bk = (unsigned)n;
            tf2x32(0u, 42u, a, bk);
            // uniform(key,(9,)): counters 0..8 padded -> pairs (i, 5+i), (4,0)
            unsigned bits[9];
            const unsigned c0[5] = {0, 1, 2, 3, 4};
            const unsigned c1[5] = {5, 6, 7, 8, 0};
            unsigned o0[5], o1[5];
            for (int i = 0; i < 5; ++i) {
                unsigned x0 = c0[i], x1 = c1[i];
                tf2x32(a, bk, x0, x1);
                o0[i] = x0; o1[i] = x1;
            }
            for (int i = 0; i < 5; ++i) bits[i] = o0[i];
            for (int i = 0; i < 4; ++i) bits[5 + i] = o1[i];
            float s2 = 0.0f;
            for (int i = 0; i < 9; ++i) {
                float f = __uint_as_float((bits[i] >> 9) | 0x3f800000u) - 1.0f;
                float l = (f - 0.5f) / 6.0f;
                if (l > 0.0f && l <  0.02f) l =  0.02f;
                if (l < 0.0f && l > -0.02f) l = -0.02f;
                if (i == 8) l = 0.0f;
                lam[n][i] = l;
                s2 += l * l;
            }
            s2s[n] = s2;
        }
    }
    __syncthreads();
    if (threadIdx.x < 4) {
        ws[S2_OFF + threadIdx.x] = s2s[threadIdx.x];
        ws[D2_OFF + threadIdx.x] = 3.0f * s2s[threadIdx.x]; // sum((2l)^2)-s2 = 3*s2
    }
    for (int idx = threadIdx.x; idx < 17 * 8 * 9; idx += blockDim.x) {
        int hi = idx / 72;
        int r  = idx % 72;
        int b  = r / 9, e = r % 9;
        float h = Hgt[b * 9 + e];
        float v = h;
        if (hi > 0) {
            int n = (hi - 1) >> 2, va = (hi - 1) & 3;
            const float coef[4] = {1.0f, 2.0f, -1.0f, -2.0f}; // +n, +2n, -n, -2n
            v = h + coef[va] * (lam[n][e] * h);
        }
        ws[HS_OFF + idx] = v;
    }
}

// =====================================================================
// Kernel C: 17 warped-SSIM partial reductions (deterministic, no atomics)
// =====================================================================
__device__ inline float bsample(const float* __restrict__ plane, float px, float py) {
    float x0f = floorf(px), y0f = floorf(py);
    int x0 = (int)x0f, y0 = (int)y0f;
    float fx = px - x0f, fy = py - y0f;
    float wa = (1.0f - fx) * (1.0f - fy);
    float wb = (1.0f - fx) * fy;
    float wc = fx * (1.0f - fy);
    float wd = fx * fy;
    float acc = 0.0f;
#pragma unroll
    for (int k = 0; k < 4; ++k) {
        int xi = x0 + (k >> 1), yi = y0 + (k & 1);
        float wgt = (k == 0) ? wa : (k == 1) ? wb : (k == 2) ? wc : wd;
        bool v = (xi >= 0) && (xi <= 127) && (yi >= 0) && (yi <= 127);
        int xc = min(max(xi, 0), 127), yc = min(max(yi, 0), 127);
        acc += wgt * (v ? plane[yc * 128 + xc] : 0.0f);
    }
    return acc;
}

__global__ __launch_bounds__(256)
void ssim_kernel(const float* __restrict__ tfm, const float* __restrict__ ifm,
                 const float* __restrict__ tw,  const float* __restrict__ iw,
                 const float* __restrict__ ws,  float* __restrict__ partials) {
    int blk = blockIdx.x;
    int hi = blk / 512;           // 512 = 8 batches * 64 chunks
    int r  = blk % 512;
    int b  = r / 64;
    int pix = (r % 64) * 256 + threadIdx.x;
    int y = pix >> 7, x = pix & 127;

    const float* H = ws + HS_OFF + (hi * 8 + b) * 9;
    float fx = (float)x, fy = (float)y;
    float inv = 1.0f / (H[6] * fx + H[7] * fy + H[8]);
    float px = (H[0] * fx + H[1] * fy + H[2]) * inv;
    float py = (H[3] * fx + H[4] * fy + H[5]) * inv;

    const float* ip = ifm + b * 16384;
    const float* wp = iw  + b * 16384;
    float wfm = bsample(ip, px, py);
    float ww  = bsample(wp, px, py);
    float d   = tfm[b * 16384 + pix] - wfm;
    float val = tw[b * 16384 + pix] * ww * d * d;

    __shared__ float red[256];
    red[threadIdx.x] = val;
    __syncthreads();
    for (int s = 128; s > 0; s >>= 1) {
        if (threadIdx.x < s) red[threadIdx.x] += red[threadIdx.x + s];
        __syncthreads();
    }
    if (threadIdx.x == 0) partials[blk] = red[0];
}

// =====================================================================
// Kernel D: final reduction + convex loss assembly
// =====================================================================
__global__ __launch_bounds__(256)
void finish_kernel(const float* __restrict__ partials, const float* __restrict__ ws,
                   float* __restrict__ out) {
    __shared__ float m[17];
    __shared__ float red[256];
    for (int hi = 0; hi < 17; ++hi) {
        float s = 0.0f;
        for (int i = threadIdx.x; i < 512; i += 256) s += partials[hi * 512 + i];
        red[threadIdx.x] = s;
        __syncthreads();
        for (int st = 128; st > 0; st >>= 1) {
            if (threadIdx.x < st) red[threadIdx.x] += red[threadIdx.x + st];
            __syncthreads();
        }
        if (threadIdx.x == 0) m[hi] = red[0] * (1.0f / 131072.0f);
        __syncthreads();
    }
    if (threadIdx.x == 0) {
        float mid = m[0];
        float cl = 0.0f;
        for (int n = 0; n < 4; ++n) {
            float s2 = ws[S2_OFF + n], d2 = ws[D2_OFF + n];
            float sL  = m[1 + 4 * n], sLL = m[2 + 4 * n];
            float sR  = m[3 + 4 * n], sRR = m[4 + 4 * n];
            cl -= fminf(sL - mid - s2, 0.0f);
            cl -= fminf(sL - sLL + d2, 0.0f);
            cl -= fminf(sR - mid - s2, 0.0f);
            cl -= fminf(sR - sRR + d2, 0.0f);
        }
        out[0] = cl;
        out[1] = mid;
    }
}

// =====================================================================
extern "C" void kernel_launch(void* const* d_in, const int* in_sizes, int n_in,
                              void* d_out, int out_size, void* d_ws, size_t ws_size,
                              hipStream_t stream) {
    const float* tf  = (const float*)d_in[0];   // template_features [8,64,128,128]
    const float* inp = (const float*)d_in[1];   // input_features
    const float* tw  = (const float*)d_in[2];   // template_weights [8,1,128,128]
    const float* iw  = (const float*)d_in[3];   // input_weights
    const float* Hgt = (const float*)d_in[4];   // H_gt [8,3,3]
    float* out = (float*)d_out;                 // [cl, mid, tfm(131072), ifm(131072)]
    float* ws  = (float*)d_ws;

    float* tfm = out + 2;
    float* ifm = out + 2 + 131072;

    fmap_kernel<<<2 * 8 * 128, 256, 0, stream>>>(tf, inp, tfm, ifm);
    setup_kernel<<<1, 256, 0, stream>>>(Hgt, ws);
    ssim_kernel<<<17 * 512, 256, 0, stream>>>(tfm, ifm, tw, iw, ws, ws + PART_OFF);
    finish_kernel<<<1, 256, 0, stream>>>(ws + PART_OFF, ws, out);
}
